// DragonHGT_46093589021066
// MI455X (gfx1250) — matrix operand order repaired and checked by reading, not compiled, not run on hardware
//
#include <hip/hip_runtime.h>
#include <hip/hip_bf16.h>

typedef __bf16 bf16_t;
typedef __attribute__((ext_vector_type(8)))  __bf16 v8bf;
typedef __attribute__((ext_vector_type(16))) __bf16 v16bf;
typedef __attribute__((ext_vector_type(8)))  float  v8f;

#define NN 50000
#define NE 200000
#define HH 4
#define DD 64
#define CC 256
#define LL 3

// ---------------------------------------------------------------------------
// Elementwise / conversion kernels
// ---------------------------------------------------------------------------

__global__ void hgt_f32_to_f32_bf16(const float* __restrict__ in,
                                    float* __restrict__ of,
                                    bf16_t* __restrict__ ob, size_t n) {
  size_t i = (size_t)blockIdx.x * blockDim.x + threadIdx.x;
  if (i >= n) return;
  float v = in[i];
  of[i] = v;
  ob[i] = (bf16_t)v;
}

__global__ void hgt_conv_copy(const float* __restrict__ in,
                              bf16_t* __restrict__ out, size_t n) {
  size_t i = (size_t)blockIdx.x * blockDim.x + threadIdx.x;
  if (i >= n) return;
  out[i] = (bf16_t)in[i];
}

// Bt[nc*K + k] = (bf16) W[k*Nc + nc]   (batched via blockIdx.y)
__global__ void hgt_conv_transpose(const float* __restrict__ W,
                                   bf16_t* __restrict__ Bt, int K, int Nc) {
  size_t tot = (size_t)K * Nc;
  size_t t = (size_t)blockIdx.x * blockDim.x + threadIdx.x;
  if (t >= tot) return;
  const float* Wz = W + (size_t)blockIdx.y * tot;
  bf16_t* Bz = Bt + (size_t)blockIdx.y * tot;
  int nc = (int)(t / K);
  int k  = (int)(t % K);
  Bz[(size_t)nc * K + k] = (bf16_t)Wz[(size_t)k * Nc + nc];
}

__global__ void hgt_fill_u32(unsigned* __restrict__ p, unsigned v, size_t n) {
  size_t i = (size_t)blockIdx.x * blockDim.x + threadIdx.x;
  if (i >= n) return;
  p[i] = v;
}

// ---------------------------------------------------------------------------
// bf16 WMMA GEMM with register blocking + LDS-staged B:
//   Y[M x Nc] = X[M x K] @ W,  W supplied as Bt[Nc x K] (bf16).
// Block = 128 threads = 4 waves. Each wave computes a 16-row x 64-col strip
// (4 v8f accumulators); the block covers 64 rows x 64 cols.
// Per 32-wide K step: each wave loads its A fragment once (reused by 4 WMMAs)
// and the block cooperatively stages the shared 64x32 B tile in LDS
// (row stride padded to 40 bf16 -> conflict-free half-wave ds_load_b128).
// grid = (ceil(M/64), Nc/64, batch).
// Epilogues: 0 = bf16 store, 1 = bias+ReLU+bf16, 2 = sigmoid-skip combine.
// ---------------------------------------------------------------------------
#define BPAD 40  // padded LDS row stride (bf16 elems) for the 32-elem K slice

__global__ __launch_bounds__(128)
void hgt_gemm_wmma(const bf16_t* __restrict__ X, int M, int ldx, long long zxs,
                   const bf16_t* __restrict__ Bt, long long zbs, int K,
                   int mode,
                   bf16_t* __restrict__ Yb, int ldy, long long zys,
                   float* __restrict__ Y32,
                   const float* __restrict__ aux, long long zaux) {
  __shared__ __align__(32) bf16_t Bsh[64 * BPAD];

  const int wave = threadIdx.x >> 5;
  const int lane = threadIdx.x & 31;
  const int half = lane >> 4;   // 0: lanes 0-15, 1: lanes 16-31
  const int l16  = lane & 15;
  const int rowBase = (blockIdx.x * 4 + wave) * 16;
  const bool active = rowBase < M;
  const int colBase = blockIdx.y * 64;
  const int z = blockIdx.z;

  const bf16_t* Xz  = X  + (size_t)z * zxs;
  const bf16_t* Btz = Bt + (size_t)z * zbs;

  const int arow = active ? (rowBase + l16) : l16;  // clamped row keeps loads legal
  // cooperative B staging assignment: thread -> (column, 16-elem K chunk)
  const int scol = threadIdx.x >> 1;
  const int soff = (threadIdx.x & 1) * 16;

  v8f acc[4] = {};

  for (int kb = 0; kb < K; kb += 32) {
    // ---- stage B tile (64 cols x 32 K, padded) into LDS ----
    {
      const v8bf* g = (const v8bf*)(Btz + (size_t)(colBase + scol) * K + kb + soff);
      v8bf b0 = g[0];
      v8bf b1 = g[1];
      *(v8bf*)(Bsh + scol * BPAD + soff)     = b0;
      *(v8bf*)(Bsh + scol * BPAD + soff + 8) = b1;
    }
    __syncthreads();

    // ---- A fragment (16x32 bf16, ISA 16-bit A layout) ----
    int ka = kb + half * 8;
    v8bf alo = *(const v8bf*)(Xz + (size_t)arow * ldx + ka);
    v8bf ahi = *(const v8bf*)(Xz + (size_t)arow * ldx + ka + 16);
    v16bf a;
#pragma unroll
    for (int i = 0; i < 8; ++i) { a[i] = alo[i]; a[i + 8] = ahi[i]; }

    // ---- 4 WMMAs reusing the A fragment across the 64-col strip ----
#pragma unroll
    for (int j = 0; j < 4; ++j) {
      int bc = j * 16 + l16;                 // column within the 64-wide tile
      v8bf blo = *(const v8bf*)(Bsh + bc * BPAD + half * 8);
      v8bf bhi = *(const v8bf*)(Bsh + bc * BPAD + half * 8 + 16);
      v16bf b;
#pragma unroll
      for (int i = 0; i < 8; ++i) { b[i] = blo[i]; b[i + 8] = bhi[i]; }
      acc[j] = __builtin_amdgcn_wmma_f32_16x16x32_bf16(false, a, false, b,
                                                       (short)0, acc[j],
                                                       false, false);
    }
    __syncthreads();
  }

  if (!active) return;

  if (mode == 0) {
    bf16_t* Yz = Yb + (size_t)z * zys;
#pragma unroll
    for (int j = 0; j < 4; ++j) {
      int col = colBase + j * 16 + l16;
#pragma unroll
      for (int r = 0; r < 8; ++r) {
        int row = rowBase + r + half * 8;
        Yz[(size_t)row * ldy + col] = (bf16_t)acc[j][r];
      }
    }
  } else if (mode == 1) {
    bf16_t* Yz = Yb + (size_t)z * zys;
#pragma unroll
    for (int j = 0; j < 4; ++j) {
      int col = colBase + j * 16 + l16;
      float bv = aux[(size_t)z * zaux + col];
#pragma unroll
      for (int r = 0; r < 8; ++r) {
        int row = rowBase + r + half * 8;
        float v = acc[j][r] + bv;
        v = v > 0.f ? v : 0.f;
        Yz[(size_t)row * ldy + col] = (bf16_t)v;
      }
    }
  } else {  // mode == 2: sigmoid-skip combine
    float be = 1.f / (1.f + __expf(-aux[0]));
#pragma unroll
    for (int j = 0; j < 4; ++j) {
      int col = colBase + j * 16 + l16;
#pragma unroll
      for (int r = 0; r < 8; ++r) {
        int row = rowBase + r + half * 8;
        size_t idx = (size_t)row * ldy + col;
        float x = be * acc[j][r] + (1.f - be) * Y32[idx];
        Y32[idx] = x;
        Yb[idx] = (bf16_t)x;
      }
    }
  }
}

// ---------------------------------------------------------------------------
// Edge phase
// ---------------------------------------------------------------------------

__device__ __forceinline__ unsigned hgt_enc(float f) {
  unsigned b = __float_as_uint(f);
  return (b & 0x80000000u) ? ~b : (b | 0x80000000u);
}
__device__ __forceinline__ float hgt_dec(unsigned u) {
  unsigned b = (u & 0x80000000u) ? (u & 0x7fffffffu) : ~u;
  return __uint_as_float(b);
}

// score[e,h] = (k[src]·qt[dst])_h * mu[h] / sqrt(D); also atomicMax per (dst,h)
__global__ void hgt_edge_score(const bf16_t* __restrict__ Kmat,
                               const bf16_t* __restrict__ Qt,
                               const int* __restrict__ src,
                               const int* __restrict__ dst,
                               const float* __restrict__ mu, float musc,
                               float* __restrict__ w,
                               unsigned* __restrict__ mbuf, int E) {
  int e = blockIdx.x * blockDim.x + threadIdx.x;
  if (e >= E) return;
  int s = src[e], d0 = dst[e];
  const v8bf* kr = (const v8bf*)(Kmat + (size_t)s * CC);
  const v8bf* qr = (const v8bf*)(Qt + (size_t)d0 * CC);
#pragma unroll
  for (int h = 0; h < HH; ++h) {
    float acc = 0.f;
#pragma unroll
    for (int j = 0; j < 8; ++j) {
      v8bf kv = kr[h * 8 + j];
      v8bf qv = qr[h * 8 + j];
#pragma unroll
      for (int t = 0; t < 8; ++t) acc += (float)kv[t] * (float)qv[t];
    }
    float sc = acc * mu[h] * musc;
    w[(size_t)e * HH + h] = sc;
    atomicMax(mbuf + (size_t)d0 * HH + h, hgt_enc(sc));
  }
}

// w[e,h] <- exp(score - max); denom[dst,h] += w
__global__ void hgt_edge_expsum(float* __restrict__ w,
                                const int* __restrict__ dst,
                                const unsigned* __restrict__ mbuf,
                                float* __restrict__ denom, int E) {
  int idx = blockIdx.x * blockDim.x + threadIdx.x;
  if (idx >= E * HH) return;
  int e = idx >> 2, h = idx & 3;
  int d0 = dst[e];
  float m = hgt_dec(mbuf[(size_t)d0 * HH + h]);
  float a = __expf(w[idx] - m);
  w[idx] = a;
  atomicAdd(denom + (size_t)d0 * HH + h, a);
}

// agg[dst, h*64+f] += w[e,h] * vm[src, h*64+f]
__global__ void hgt_edge_scatter(const float* __restrict__ w,
                                 const int* __restrict__ src,
                                 const int* __restrict__ dst,
                                 const bf16_t* __restrict__ vm,
                                 float* __restrict__ agg, int E) {
  int idx = blockIdx.x * blockDim.x + threadIdx.x;
  if (idx >= E * DD) return;
  int e = idx >> 6, f = idx & 63;
  int s = src[e], d0 = dst[e];
  const bf16_t* vr = vm + (size_t)s * CC + f;
  float* ar = agg + (size_t)d0 * CC + f;
  const float* we = w + (size_t)e * HH;
#pragma unroll
  for (int h = 0; h < HH; ++h)
    atomicAdd(ar + h * DD, we[h] * (float)vr[h * DD]);
}

// hb = bf16( gelu_exact( agg / (denom + 1e-16) ) )
__global__ void hgt_norm_gelu(const float* __restrict__ agg,
                              const float* __restrict__ denom,
                              bf16_t* __restrict__ hb, size_t n) {
  size_t i = (size_t)blockIdx.x * blockDim.x + threadIdx.x;
  if (i >= n) return;
  size_t nd = i >> 8;
  int c = (int)(i & 255);
  float v = agg[i] / (denom[nd * HH + (c >> 6)] + 1e-16f);
  float g = 0.5f * v * (1.f + erff(v * 0.70710678118654752f));
  hb[i] = (bf16_t)g;
}

// Per-head final projection [N,128] x [128,8] (+bias), write sliced columns.
__global__ void hgt_head_out(const bf16_t* __restrict__ tb,
                             const float* __restrict__ W2,
                             const float* __restrict__ b2,
                             float* __restrict__ out, int off, int width) {
  int n = blockIdx.x * blockDim.x + threadIdx.x;
  if (n >= NN) return;
  float acc[8];
#pragma unroll
  for (int o = 0; o < 8; ++o) acc[o] = b2[o];
  const v8bf* tr = (const v8bf*)(tb + (size_t)n * 128);
#pragma unroll 2
  for (int j = 0; j < 16; ++j) {
    v8bf tv = tr[j];
#pragma unroll
    for (int t = 0; t < 8; ++t) {
      float x = (float)tv[t];
      const float* wr = W2 + (size_t)(j * 8 + t) * 8;
#pragma unroll
      for (int o = 0; o < 8; ++o) acc[o] += x * wr[o];
    }
  }
  for (int o = 0; o < width; ++o) out[(size_t)n * 24 + off + o] = acc[o];
}

// ---------------------------------------------------------------------------
// Host orchestration
// ---------------------------------------------------------------------------
extern "C" void kernel_launch(void* const* d_in, const int* in_sizes, int n_in,
                              void* d_out, int out_size, void* d_ws, size_t ws_size,
                              hipStream_t stream) {
  (void)in_sizes; (void)n_in; (void)out_size; (void)ws_size;
  const float* xe_in = (const float*)d_in[0];
  const float* xs_in = (const float*)d_in[1];
  const int* ee_src = (const int*)d_in[2];
  const int* ee_dst = (const int*)d_in[3];
  const int* es_src = (const int*)d_in[4];
  const int* es_dst = (const int*)d_in[5];
  const int* se_src = (const int*)d_in[6];
  const int* se_dst = (const int*)d_in[7];
  const float* Wk   = (const float*)d_in[8];
  const float* Wq   = (const float*)d_in[9];
  const float* Wv   = (const float*)d_in[10];
  const float* Wa   = (const float*)d_in[11];
  const float* skip = (const float*)d_in[12];
  const float* Arel = (const float*)d_in[13];
  const float* Mrel = (const float*)d_in[14];
  const float* mu   = (const float*)d_in[15];
  const float* Wsh  = (const float*)d_in[16];
  const float* bsh  = (const float*)d_in[17];
  const float* Wt1  = (const float*)d_in[18];
  const float* bt1  = (const float*)d_in[19];
  const float* Wt2  = (const float*)d_in[20];
  const float* bt2  = (const float*)d_in[21];
  float* out = (float*)d_out;

  char* wp = (char*)d_ws;
  auto alloc = [&](size_t bytes) -> void* {
    void* r = (void*)wp;
    wp += (bytes + 255) & ~(size_t)255;
    return r;
  };
  const size_t NC = (size_t)NN * CC;

  float*  xe_f = (float*)alloc(NC * 4);
  float*  xs_f = (float*)alloc(NC * 4);
  bf16_t* xeb  = (bf16_t*)alloc(NC * 2);
  bf16_t* xsb  = (bf16_t*)alloc(NC * 2);
  bf16_t* keb  = (bf16_t*)alloc(NC * 2);
  bf16_t* qeb  = (bf16_t*)alloc(NC * 2);
  bf16_t* veb  = (bf16_t*)alloc(NC * 2);
  bf16_t* ksb  = (bf16_t*)alloc(NC * 2);
  bf16_t* qsb  = (bf16_t*)alloc(NC * 2);
  bf16_t* vsb  = (bf16_t*)alloc(NC * 2);
  bf16_t* qtb  = (bf16_t*)alloc(NC * 2);
  bf16_t* vmb  = (bf16_t*)alloc(NC * 2);
  bf16_t* hb   = (bf16_t*)alloc(NC * 2);
  bf16_t* hb2  = (bf16_t*)alloc(NC * 2);
  bf16_t* tbuf = (bf16_t*)alloc((size_t)NN * 128 * 2);
  float*  agg  = (float*)alloc(NC * 4);
  unsigned* mbuf = (unsigned*)alloc((size_t)NN * HH * 4);
  float*  denom  = (float*)alloc((size_t)NN * HH * 4);
  float*  w0 = (float*)alloc((size_t)NE * HH * 4);
  float*  w1 = (float*)alloc((size_t)NE * HH * 4);
  float*  w2 = (float*)alloc((size_t)NE * HH * 4);
  // bf16 weights (pre-transposed into Bt[Nc][K] layout)
  const size_t WSZ = (size_t)CC * CC;           // 65536
  const size_t RSZ = (size_t)HH * DD * DD;      // 16384
  bf16_t* WkT  = (bf16_t*)alloc(LL * 2 * WSZ * 2);
  bf16_t* WqT  = (bf16_t*)alloc(LL * 2 * WSZ * 2);
  bf16_t* WvT  = (bf16_t*)alloc(LL * 2 * WSZ * 2);
  bf16_t* WaT  = (bf16_t*)alloc(LL * 2 * WSZ * 2);
  bf16_t* AB   = (bf16_t*)alloc(LL * 3 * RSZ * 2);  // A used as Bt directly
  bf16_t* MT   = (bf16_t*)alloc(LL * 3 * RSZ * 2);  // M transposed per head
  bf16_t* WshT = (bf16_t*)alloc(2 * WSZ * 2);
  bf16_t* Wt1T = (bf16_t*)alloc((size_t)7 * 128 * 256 * 2);

  const int GX = (NN / 16 + 3) / 4;  // 782 blocks of 4 row tiles (last clamps)
  auto gemm = [&](const bf16_t* X, int ldx, long long zxs,
                  const bf16_t* Bt, long long zbs, int K, int Nc, int batch,
                  int mode, bf16_t* Yb, int ldy, long long zys,
                  float* Y32, const float* aux, long long zaux) {
    hgt_gemm_wmma<<<dim3(GX, Nc / 64, batch), 128, 0, stream>>>(
        X, NN, ldx, zxs, Bt, zbs, K, mode, Yb, ldy, zys, Y32, aux, zaux);
  };
  auto fill0 = [&](void* ptr, size_t nelem) {
    hgt_fill_u32<<<dim3((unsigned)((nelem + 255) / 256)), 256, 0, stream>>>(
        (unsigned*)ptr, 0u, nelem);
  };

  // ---- setup: inputs + weight conversion ----
  hgt_f32_to_f32_bf16<<<dim3((unsigned)((NC + 255) / 256)), 256, 0, stream>>>(xe_in, xe_f, xeb, NC);
  hgt_f32_to_f32_bf16<<<dim3((unsigned)((NC + 255) / 256)), 256, 0, stream>>>(xs_in, xs_f, xsb, NC);
  hgt_conv_transpose<<<dim3(256, LL * 2), 256, 0, stream>>>(Wk, WkT, CC, CC);
  hgt_conv_transpose<<<dim3(256, LL * 2), 256, 0, stream>>>(Wq, WqT, CC, CC);
  hgt_conv_transpose<<<dim3(256, LL * 2), 256, 0, stream>>>(Wv, WvT, CC, CC);
  hgt_conv_transpose<<<dim3(256, LL * 2), 256, 0, stream>>>(Wa, WaT, CC, CC);
  hgt_conv_copy<<<dim3((unsigned)((LL * 3 * RSZ + 255) / 256)), 256, 0, stream>>>(Arel, AB, LL * 3 * RSZ);
  hgt_conv_transpose<<<dim3(16, LL * 3 * HH), 256, 0, stream>>>(Mrel, MT, DD, DD);
  hgt_conv_transpose<<<dim3(256, 2), 256, 0, stream>>>(Wsh, WshT, CC, CC);
  hgt_conv_transpose<<<dim3(128, 7), 256, 0, stream>>>(Wt1, Wt1T, CC, 128);

  const float musc = 0.125f;  // 1/sqrt(64)
  dim3 gE((NE + 255) / 256), bE(256);
  dim3 gEH((NE * HH + 255) / 256);
  dim3 gEF((NE * DD + 255) / 256);
  dim3 gNC((unsigned)((NC + 255) / 256));

  for (int l = 0; l < LL; ++l) {
    // node projections K/Q/V for both types (bf16 WMMA GEMMs)
    gemm(xeb, CC, 0, WkT + (l * 2 + 0) * WSZ, 0, CC, CC, 1, 0, keb, CC, 0, nullptr, nullptr, 0);
    gemm(xeb, CC, 0, WqT + (l * 2 + 0) * WSZ, 0, CC, CC, 1, 0, qeb, CC, 0, nullptr, nullptr, 0);
    gemm(xeb, CC, 0, WvT + (l * 2 + 0) * WSZ, 0, CC, CC, 1, 0, veb, CC, 0, nullptr, nullptr, 0);
    gemm(xsb, CC, 0, WkT + (l * 2 + 1) * WSZ, 0, CC, CC, 1, 0, ksb, CC, 0, nullptr, nullptr, 0);
    gemm(xsb, CC, 0, WqT + (l * 2 + 1) * WSZ, 0, CC, CC, 1, 0, qsb, CC, 0, nullptr, nullptr, 0);
    gemm(xsb, CC, 0, WvT + (l * 2 + 1) * WSZ, 0, CC, CC, 1, 0, vsb, CC, 0, nullptr, nullptr, 0);

    // ---- destination 'stmt' (relation 1: expr->stmt) ----
    fill0(mbuf, (size_t)NN * HH);
    fill0(denom, (size_t)NN * HH);
    fill0(agg, NC);
    // qt[n,h,:] = qs[n,h,:] @ A1[h]^T  (Bt = A1[h] as stored), batched per head
    gemm(qsb, CC, DD, AB + (l * 3 + 1) * RSZ, DD * DD, DD, DD, HH, 0, qtb, CC, DD, nullptr, nullptr, 0);
    hgt_edge_score<<<gE, bE, 0, stream>>>(keb, qtb, es_src, es_dst,
                                          mu + (l * 3 + 1) * HH, musc, w1, mbuf, NE);
    hgt_edge_expsum<<<gEH, bE, 0, stream>>>(w1, es_dst, mbuf, denom, NE);
    // vm[n,h,:] = ve[n,h,:] @ M1[h]  (Bt = M1[h]^T)
    gemm(veb, CC, DD, MT + (l * 3 + 1) * RSZ, DD * DD, DD, DD, HH, 0, vmb, CC, DD, nullptr, nullptr, 0);
    hgt_edge_scatter<<<gEF, bE, 0, stream>>>(w1, es_src, es_dst, vmb, agg, NE);
    hgt_norm_gelu<<<gNC, 256, 0, stream>>>(agg, denom, hb, NC);
    // xs = sigmoid(skip)*gelu(agg)@Wa + (1-sigmoid)*xs   (skip epilogue)
    gemm(hb, CC, 0, WaT + (l * 2 + 1) * WSZ, 0, CC, CC, 1, 2, xsb, CC, 0, xs_f, skip + l * 2 + 1, 0);

    // ---- destination 'expr' (relations 0: expr->expr and 2: stmt->expr) ----
    fill0(mbuf, (size_t)NN * HH);
    fill0(denom, (size_t)NN * HH);
    fill0(agg, NC);
    gemm(qeb, CC, DD, AB + (l * 3 + 0) * RSZ, DD * DD, DD, DD, HH, 0, qtb, CC, DD, nullptr, nullptr, 0);
    hgt_edge_score<<<gE, bE, 0, stream>>>(keb, qtb, ee_src, ee_dst,
                                          mu + (l * 3 + 0) * HH, musc, w0, mbuf, NE);
    gemm(qeb, CC, DD, AB + (l * 3 + 2) * RSZ, DD * DD, DD, DD, HH, 0, qtb, CC, DD, nullptr, nullptr, 0);
    hgt_edge_score<<<gE, bE, 0, stream>>>(ksb, qtb, se_src, se_dst,
                                          mu + (l * 3 + 2) * HH, musc, w2, mbuf, NE);
    hgt_edge_expsum<<<gEH, bE, 0, stream>>>(w0, ee_dst, mbuf, denom, NE);
    hgt_edge_expsum<<<gEH, bE, 0, stream>>>(w2, se_dst, mbuf, denom, NE);
    gemm(veb, CC, DD, MT + (l * 3 + 0) * RSZ, DD * DD, DD, DD, HH, 0, vmb, CC, DD, nullptr, nullptr, 0);
    hgt_edge_scatter<<<gEF, bE, 0, stream>>>(w0, ee_src, ee_dst, vmb, agg, NE);
    gemm(vsb, CC, DD, MT + (l * 3 + 2) * RSZ, DD * DD, DD, DD, HH, 0, vmb, CC, DD, nullptr, nullptr, 0);
    hgt_edge_scatter<<<gEF, bE, 0, stream>>>(w2, se_src, se_dst, vmb, agg, NE);
    hgt_norm_gelu<<<gNC, 256, 0, stream>>>(agg, denom, hb, NC);
    gemm(hb, CC, 0, WaT + (l * 2 + 0) * WSZ, 0, CC, CC, 1, 2, xeb, CC, 0, xe_f, skip + l * 2 + 0, 0);
  }

  // ---- shared MLP on 'expr' nodes (bias+ReLU epilogue) ----
  gemm(xeb, CC, 0, WshT + 0 * WSZ, 0, CC, CC, 1, 1, hb,  CC, 0, nullptr, bsh + 0,  0);
  gemm(hb,  CC, 0, WshT + 1 * WSZ, 0, CC, CC, 1, 1, hb2, CC, 0, nullptr, bsh + CC, 0);

  // ---- 7 task heads ----
  const int offs[7] = {0, 3, 6, 9, 14, 15, 16};
  const int dims[7] = {3, 3, 3, 5, 1, 1, 8};
  for (int k = 0; k < 7; ++k) {
    gemm(hb2, CC, 0, Wt1T + (size_t)k * 128 * 256, 0, CC, 128, 1, 1,
         tbuf, 128, 0, nullptr, bt1 + k * 128, 0);
    hgt_head_out<<<dim3((NN + 255) / 256), 256, 0, stream>>>(
        tbuf, Wt2 + (size_t)k * 128 * 8, bt2 + k * 8, out, offs[k], dims[k]);
  }
}
